// GlobalSceneEncoder_83442624627330
// MI455X (gfx1250) — matrix operand order repaired and checked by reading, not compiled, not run on hardware
//
#include <hip/hip_runtime.h>
#include <hip/hip_bf16.h>
#include <stdint.h>
#include <stddef.h>

typedef __attribute__((ext_vector_type(16))) __bf16 v16bf;
typedef __attribute__((ext_vector_type(8)))  float  v8f;
typedef __attribute__((ext_vector_type(4)))  int    v4i;
typedef __bf16 bf16_t;

#define D_DIM  384
#define KNN_K  16
#define N0     4096
#define FPS_T  256
#define BLK_ROWS 128
#define BLK_COLS 64

#ifndef __has_builtin
#define __has_builtin(x) 0
#endif
#if __has_builtin(__builtin_amdgcn_global_load_async_to_lds_b128)
#define USE_ASYNC_LDS 1
#else
#define USE_ASYNC_LDS 0
#endif

// async-to-LDS builtin takes (global v4i*, lds v4i*, imm offset, imm cpol)
typedef __attribute__((address_space(1))) v4i* g_v4i_p;
typedef __attribute__((address_space(3))) v4i* l_v4i_p;

// ---------------------------------------------------------------------------
// Weight convert + transpose:  W fp32 [K x N]  ->  WT bf16 [N x K]
// so each lane's B-fragment (16 contiguous K values of one column) is two
// contiguous 16B reads, and a 64-column slab is one contiguous 48KB region.
// ---------------------------------------------------------------------------
__global__ void wcvt_kernel(const float* __restrict__ W, bf16_t* __restrict__ WT,
                            int K, int N)
{
    int i = blockIdx.x * blockDim.x + threadIdx.x;
    if (i >= K * N) return;
    int k = i / N, n = i - k * N;
    WT[(size_t)n * K + k] = (bf16_t)W[i];
}

// ---------------------------------------------------------------------------
// WMMA GEMM: C = act(A * B + bias). Block = 8 waves = 128 rows x 64 cols.
// The bf16 B slab [64 x K] is staged once into LDS with
// global_load_async_to_lds_b128 (ASYNCcnt) and shared by all waves. Each wave
// computes a 16x64 strip: one A fragment feeds 4 v_wmma_f32_16x16x32_bf16.
// D layout per CDNA5 ISA 7.12.2 (wave32); A/B use contiguous k = 16h + s.
// ---------------------------------------------------------------------------
template <bool RELU, bool GATHER>
__global__ void __launch_bounds__(256)
wmma_gemm_bf16_kernel(const float* __restrict__ A,
                      const bf16_t* __restrict__ BT,
                      const float* __restrict__ bias,
                      const int* __restrict__ ridx,
                      float* __restrict__ C,
                      int M, int N, int K)   // K == D_DIM
{
    __shared__ bf16_t ldsB[BLK_COLS * D_DIM];          // 48 KB
    const int colBlocks = N / BLK_COLS;
    const int bM = blockIdx.x / colBlocks;
    const int bN = (blockIdx.x - bM * colBlocks) * BLK_COLS;

    // ---- cooperative staging of the B slab (contiguous 48KB) ----
    {
        const int nChunks = (BLK_COLS * D_DIM * 2) / 16;   // 16B chunks
        const char* g = (const char*)(BT + (size_t)bN * K);
        char* l = (char*)ldsB;
#if USE_ASYNC_LDS
        for (int c = threadIdx.x; c < nChunks; c += 256) {
            __builtin_amdgcn_global_load_async_to_lds_b128(
                (g_v4i_p)(g + (size_t)c * 16),
                (l_v4i_p)(l + (size_t)c * 16), 0, 0);
        }
#if __has_builtin(__builtin_amdgcn_s_wait_asynccnt)
        __builtin_amdgcn_s_wait_asynccnt(0);
#else
        asm volatile("s_wait_asynccnt 0" ::: "memory");
#endif
#else
        for (int c = threadIdx.x; c < nChunks; c += 256) {
            ((int4*)l)[c] = ((const int4*)g)[c];
        }
#endif
        __syncthreads();
    }

    const int lane = threadIdx.x & 31;
    const int wv   = threadIdx.x >> 5;
    const int tM   = bM * BLK_ROWS + wv * 16;
    if (tM >= M) return;   // inactive wave (no barriers beyond this point)

    const int hs  = lane >> 4;
    const int l15 = lane & 15;

    int arow = tM + l15;
    if (GATHER) arow = ridx[arow];
    const float* Ap = A + (size_t)arow * K;

    v8f acc[4];
    #pragma unroll
    for (int t = 0; t < 4; ++t) {
        const float bv = bias ? bias[bN + t * 16 + l15] : 0.0f;
        #pragma unroll
        for (int j = 0; j < 8; ++j) acc[t][j] = bv;
    }

    for (int kb = 0; kb < K; kb += 32) {
        // A 16-bit 16x32 layout (contiguous, matches B): lanes0-15 K{0..15};
        // lanes16-31 K{16..31}.
        v16bf af;
        #pragma unroll
        for (int g = 0; g < 2; ++g) {
            #pragma unroll
            for (int p = 0; p < 4; ++p) {
                const int k = kb + hs * 16 + g * 8 + p * 2;
                const int e = g * 8 + p * 2;
                af[e]     = (bf16_t)Ap[k];
                af[e + 1] = (bf16_t)Ap[k + 1];
            }
        }
        // 4 B fragments from LDS (16 contiguous bf16 = 2x ds_load_b128 each)
        #pragma unroll
        for (int t = 0; t < 4; ++t) {
            const bf16_t* lb = ldsB + (size_t)(t * 16 + l15) * K + kb + hs * 16;
            v16bf bfr;
            #pragma unroll
            for (int e = 0; e < 16; ++e) bfr[e] = lb[e];
            acc[t] = __builtin_amdgcn_wmma_f32_16x16x32_bf16(false, af, false, bfr,
                                                             (short)0, acc[t],
                                                             false, false);
        }
    }

    // D layout: lanes0-15 rows tM+0..7, lanes16-31 rows tM+8..15.
    const int rb = tM + hs * 8;
    #pragma unroll
    for (int t = 0; t < 4; ++t) {
        const int col = bN + t * 16 + l15;
        #pragma unroll
        for (int j = 0; j < 8; ++j) {
            float v = acc[t][j];
            if (RELU) v = fmaxf(v, 0.0f);
            C[(size_t)(rb + j) * N + col] = v;
        }
    }
}

static void run_gemm(const float* A, const bf16_t* BT, const float* bias,
                     const int* ridx, float* C, int M, int N, int K,
                     bool relu, hipStream_t stream)
{
    const int blocks = ((M + BLK_ROWS - 1) / BLK_ROWS) * (N / BLK_COLS);
    if (relu) {
        if (ridx) wmma_gemm_bf16_kernel<true , true ><<<blocks, 256, 0, stream>>>(A, BT, bias, ridx, C, M, N, K);
        else      wmma_gemm_bf16_kernel<true , false><<<blocks, 256, 0, stream>>>(A, BT, bias, ridx, C, M, N, K);
    } else {
        if (ridx) wmma_gemm_bf16_kernel<false, true ><<<blocks, 256, 0, stream>>>(A, BT, bias, ridx, C, M, N, K);
        else      wmma_gemm_bf16_kernel<false, false><<<blocks, 256, 0, stream>>>(A, BT, bias, ridx, C, M, N, K);
    }
}

// ---------------------------------------------------------------------------
// Farthest point sampling: one workgroup, sequential M-1 argmax steps,
// per-thread dmin in registers, LDS tree reduction for the argmax.
// ---------------------------------------------------------------------------
__global__ void fps_kernel(const float* __restrict__ pos, int N, int M,
                           int* __restrict__ idx)
{
    __shared__ float svals[FPS_T];
    __shared__ int   sinds[FPS_T];
    __shared__ float scur[3];
    const int t   = threadIdx.x;
    const int per = (N + FPS_T - 1) / FPS_T;   // <= 16

    const float px0 = pos[0], py0 = pos[1], pz0 = pos[2];
    float dmin[16];
    for (int j = 0; j < per; ++j) {
        const int i = t + j * FPS_T;
        if (i < N) {
            const float dx = pos[3*i] - px0, dy = pos[3*i+1] - py0, dz = pos[3*i+2] - pz0;
            dmin[j] = dx*dx + dy*dy + dz*dz;
        } else dmin[j] = -1.0f;
    }
    if (t == 0) idx[0] = 0;

    for (int s = 1; s < M; ++s) {
        float bv = -1.0f; int bi = 0;
        for (int j = 0; j < per; ++j) {
            const int i = t + j * FPS_T;
            if (i < N && dmin[j] > bv) { bv = dmin[j]; bi = i; }
        }
        svals[t] = bv; sinds[t] = bi;
        __syncthreads();
        for (int off = FPS_T / 2; off > 0; off >>= 1) {
            if (t < off && svals[t + off] > svals[t]) {
                svals[t] = svals[t + off]; sinds[t] = sinds[t + off];
            }
            __syncthreads();
        }
        if (t == 0) {
            const int nxt = sinds[0];
            idx[s]  = nxt;
            scur[0] = pos[3*nxt]; scur[1] = pos[3*nxt+1]; scur[2] = pos[3*nxt+2];
        }
        __syncthreads();
        const float qx = scur[0], qy = scur[1], qz = scur[2];
        for (int j = 0; j < per; ++j) {
            const int i = t + j * FPS_T;
            if (i < N) {
                const float dx = pos[3*i] - qx, dy = pos[3*i+1] - qy, dz = pos[3*i+2] - qz;
                dmin[j] = fminf(dmin[j], dx*dx + dy*dy + dz*dz);
            }
        }
        __syncthreads();
    }
}

__global__ void gather_pos_kernel(const float* __restrict__ pos,
                                  const int* __restrict__ idx,
                                  float* __restrict__ qpos, int M)
{
    int i = blockIdx.x * blockDim.x + threadIdx.x;
    if (i >= M * 3) return;
    const int m = i / 3, c = i - m * 3;
    qpos[i] = pos[idx[m] * 3 + c];
}

// ---------------------------------------------------------------------------
// kNN (K=16 smallest d2): one wave per query; per-lane register top-K, then
// 16 rounds of wave-wide min via __shfl (wave32).
// ---------------------------------------------------------------------------
__global__ void knn_kernel(const float* __restrict__ qpos, const float* __restrict__ pos,
                           int M, int N, int* __restrict__ nbr, float* __restrict__ nd2)
{
    const int lane = threadIdx.x & 31;
    const int wv   = threadIdx.x >> 5;
    const int q    = blockIdx.x * (blockDim.x >> 5) + wv;
    if (q >= M) return;
    const float qx = qpos[3*q], qy = qpos[3*q+1], qz = qpos[3*q+2];

    float bd[KNN_K]; int bi[KNN_K];
    #pragma unroll
    for (int j = 0; j < KNN_K; ++j) { bd[j] = 3.0e38f; bi[j] = 0; }
    float worstv = 3.0e38f; int worst = 0;

    for (int i = lane; i < N; i += 32) {
        const float dx = pos[3*i] - qx, dy = pos[3*i+1] - qy, dz = pos[3*i+2] - qz;
        const float d = dx*dx + dy*dy + dz*dz;
        if (d < worstv) {
            #pragma unroll
            for (int j = 0; j < KNN_K; ++j) if (j == worst) { bd[j] = d; bi[j] = i; }
            worstv = -1.0f;
            #pragma unroll
            for (int j = 0; j < KNN_K; ++j) if (bd[j] > worstv) { worstv = bd[j]; worst = j; }
        }
    }
    for (int sel = 0; sel < KNN_K; ++sel) {
        float mv = 3.0e38f; int ms = 0;
        #pragma unroll
        for (int j = 0; j < KNN_K; ++j) if (bd[j] < mv) { mv = bd[j]; ms = j; }
        float rv = mv; int rl = lane;
        #pragma unroll
        for (int off = 16; off > 0; off >>= 1) {
            const float ov = __shfl_down(rv, off, 32);
            const int   ol = __shfl_down(rl, off, 32);
            if (ov < rv) { rv = ov; rl = ol; }
        }
        rl = __shfl(rl, 0, 32);
        if (lane == rl) {
            nbr[q * KNN_K + sel] = bi[ms];
            nd2[q * KNN_K + sel] = mv;
            #pragma unroll
            for (int j = 0; j < KNN_K; ++j) if (j == ms) bd[j] = 3.0e38f;
        }
    }
}

// hidden = relu(rel @ pw1 + pb1), rel = q_pos[m] - pos[nbr[mk]]  (inner dim 3)
__global__ void pos_mlp1_kernel(const float* __restrict__ qpos, const float* __restrict__ pos,
                                const int* __restrict__ nbr,
                                const float* __restrict__ pw1, const float* __restrict__ pb1,
                                float* __restrict__ hidden, int M, int D)
{
    int i = blockIdx.x * blockDim.x + threadIdx.x;
    if (i >= M * KNN_K * D) return;
    const int d  = i % D;
    const int mk = i / D;
    const int m  = mk / KNN_K;
    const int nb = nbr[mk];
    const float rx = qpos[3*m]   - pos[3*nb];
    const float ry = qpos[3*m+1] - pos[3*nb+1];
    const float rz = qpos[3*m+2] - pos[3*nb+2];
    const float h = rx * pw1[d] + ry * pw1[D + d] + rz * pw1[2*D + d] + pb1[d];
    hidden[i] = fmaxf(h, 0.0f);
}

// t = a_dst[m] - a_src[nbr[mk]] + delta[mk]
__global__ void attn_input_kernel(const float* __restrict__ adst, const float* __restrict__ asrc,
                                  const float* __restrict__ delta, const int* __restrict__ nbr,
                                  float* __restrict__ t, int M, int D)
{
    int i = blockIdx.x * blockDim.x + threadIdx.x;
    if (i >= M * KNN_K * D) return;
    const int d  = i % D;
    const int mk = i / D;
    const int m  = mk / KNN_K;
    const int nb = nbr[mk];
    t[i] = adst[(size_t)m * D + d] - asrc[(size_t)nb * D + d] + delta[i];
}

// masked softmax over K per (m,d), then out = sum attn * (v[nbr] + delta)
__global__ void softmax_combine_kernel(const float* __restrict__ alpha, const float* __restrict__ vfeat,
                                       const float* __restrict__ delta, const int* __restrict__ nbr,
                                       const float* __restrict__ nd2, float r2,
                                       float* __restrict__ out, int M, int D)
{
    int i = blockIdx.x * blockDim.x + threadIdx.x;
    if (i >= M * D) return;
    const int d = i % D;
    const int m = i / D;
    float av[KNN_K]; bool vl[KNN_K];
    float mx = -3.0e38f;
    #pragma unroll
    for (int k = 0; k < KNN_K; ++k) {
        const int mk = m * KNN_K + k;
        const bool valid = nd2[mk] <= r2;
        const float a = valid ? alpha[(size_t)mk * D + d] : -1.0e30f;
        av[k] = a; vl[k] = valid;
        mx = fmaxf(mx, a);
    }
    float ssum = 0.0f;
    #pragma unroll
    for (int k = 0; k < KNN_K; ++k) { av[k] = __expf(av[k] - mx); ssum += av[k]; }
    const float inv = 1.0f / ssum;
    float o = 0.0f;
    #pragma unroll
    for (int k = 0; k < KNN_K; ++k) {
        const int mk = m * KNN_K + k;
        if (vl[k]) o += av[k] * inv * (vfeat[(size_t)nbr[mk] * D + d] + delta[(size_t)mk * D + d]);
    }
    out[i] = o;
}

// in-place layernorm over D per row; one wave per row, shfl reductions
__global__ void layernorm_kernel(float* __restrict__ x, const float* __restrict__ g,
                                 const float* __restrict__ b, int R, int D)
{
    const int lane = threadIdx.x & 31;
    const int wv   = threadIdx.x >> 5;
    const int row  = blockIdx.x * (blockDim.x >> 5) + wv;
    if (row >= R) return;
    float s = 0.0f, s2 = 0.0f;
    for (int d = lane; d < D; d += 32) {
        const float v = x[(size_t)row * D + d];
        s += v; s2 += v * v;
    }
    #pragma unroll
    for (int off = 16; off > 0; off >>= 1) {
        s  += __shfl_down(s,  off, 32);
        s2 += __shfl_down(s2, off, 32);
    }
    s = __shfl(s, 0, 32); s2 = __shfl(s2, 0, 32);
    const float mean = s / D;
    const float var  = s2 / D - mean * mean;
    const float inv  = rsqrtf(var + 1e-5f);
    for (int d = lane; d < D; d += 32) {
        const size_t o = (size_t)row * D + d;
        x[o] = (x[o] - mean) * inv * g[d] + b[d];
    }
}

__global__ void extract_kernel(const float* __restrict__ pts, float* __restrict__ pos,
                               float* __restrict__ x, int N, int D)
{
    int i = blockIdx.x * blockDim.x + threadIdx.x;
    const int C = 3 + D;
    if (i >= N * C) return;
    const int n = i / C, c = i - n * C;
    const float v = pts[i];
    if (c < 3) pos[n * 3 + c] = v;
    else       x[(size_t)n * D + (c - 3)] = v;
}

__global__ void copy_kernel(const float* __restrict__ src, float* __restrict__ dst, int n)
{
    int i = blockIdx.x * blockDim.x + threadIdx.x;
    if (i < n) dst[i] = src[i];
}

__global__ void zero_kernel(float* __restrict__ p, int n)
{
    int i = blockIdx.x * blockDim.x + threadIdx.x;
    if (i < n) p[i] = 0.0f;
}

// ---------------------------------------------------------------------------
extern "C" void kernel_launch(void* const* d_in, const int* in_sizes, int n_in,
                              void* d_out, int out_size, void* d_ws, size_t ws_size,
                              hipStream_t stream)
{
    (void)n_in; (void)out_size; (void)ws_size;
    const int D = D_DIM;
    const float* pts    = (const float*)d_in[0];
    const float* lin_w  = (const float*)d_in[1];
    const float* lin_b  = (const float*)d_in[2];
    const float* src_w  = (const float*)d_in[3];
    const float* dst_w  = (const float*)d_in[4];
    const float* pos_w1 = (const float*)d_in[5];
    const float* pos_b1 = (const float*)d_in[6];
    const float* pos_w2 = (const float*)d_in[7];
    const float* pos_b2 = (const float*)d_in[8];
    const float* attn_w = (const float*)d_in[9];
    const float* attn_b = (const float*)d_in[10];
    const float* ln_g   = (const float*)d_in[11];
    const float* ln_b   = (const float*)d_in[12];
    const float* proj_w = (const float*)d_in[13];
    const float* proj_b = (const float*)d_in[14];
    const float* pln_g  = (const float*)d_in[15];
    const float* pln_b  = (const float*)d_in[16];

    const int B = in_sizes[0] / (N0 * (3 + D));
    const float radii[4] = {1.0f, 2.0f, 4.0f, 8.0f};

    // ---- workspace carve (per-batch reuse; ~110 MB total) ----
    char* base = (char*)d_ws;
    size_t off = 0;
    auto carve = [&](size_t bytes) -> char* {
        char* p = base + off;
        off += (bytes + 255) & ~(size_t)255;
        return p;
    };
    const size_t matb = (size_t)D * D * sizeof(bf16_t);
    bf16_t *wt_lin[4], *wt_src[4], *wt_dst[4], *wt_pw2[4], *wt_attn[4];
    for (int s = 0; s < 4; ++s) wt_lin[s]  = (bf16_t*)carve(matb);
    for (int s = 0; s < 4; ++s) wt_src[s]  = (bf16_t*)carve(matb);
    for (int s = 0; s < 4; ++s) wt_dst[s]  = (bf16_t*)carve(matb);
    for (int s = 0; s < 4; ++s) wt_pw2[s]  = (bf16_t*)carve(matb);
    for (int s = 0; s < 4; ++s) wt_attn[s] = (bf16_t*)carve(matb);
    bf16_t* wt_proj = (bf16_t*)carve(matb);

    float* xA     = (float*)carve((size_t)N0 * D * sizeof(float));
    float* xB     = (float*)carve((size_t)N0 * D * sizeof(float));
    float* posA   = (float*)carve((size_t)N0 * 3 * sizeof(float));
    float* posB   = (float*)carve((size_t)N0 * 3 * sizeof(float));
    int*   idx    = (int*)  carve(1024 * sizeof(int));
    int*   nbr    = (int*)  carve(16384 * sizeof(int));
    float* nd2    = (float*)carve(16384 * sizeof(float));
    float* vbuf   = (float*)carve((size_t)N0 * D * sizeof(float));
    float* asrc   = (float*)carve((size_t)N0 * D * sizeof(float));
    float* adst   = (float*)carve((size_t)1024 * D * sizeof(float));
    float* hidden = (float*)carve((size_t)16384 * D * sizeof(float)); // reused as attn input
    float* delta  = (float*)carve((size_t)16384 * D * sizeof(float));
    float* alpha  = (float*)carve((size_t)16384 * D * sizeof(float));
    float* featb  = (float*)carve((size_t)16 * D * sizeof(float));

    // ---- convert+transpose all weights to bf16 [N][K] (deterministic, every call) ----
    {
        dim3 blk(256), grd((D * D + 255) / 256);
        for (int s = 0; s < 4; ++s) {
            wcvt_kernel<<<grd, blk, 0, stream>>>(lin_w  + (size_t)s * D * D, wt_lin[s],  D, D);
            wcvt_kernel<<<grd, blk, 0, stream>>>(src_w  + (size_t)s * D * D, wt_src[s],  D, D);
            wcvt_kernel<<<grd, blk, 0, stream>>>(dst_w  + (size_t)s * D * D, wt_dst[s],  D, D);
            wcvt_kernel<<<grd, blk, 0, stream>>>(pos_w2 + (size_t)s * D * D, wt_pw2[s],  D, D);
            wcvt_kernel<<<grd, blk, 0, stream>>>(attn_w + (size_t)s * D * D, wt_attn[s], D, D);
        }
        wcvt_kernel<<<grd, blk, 0, stream>>>(proj_w, wt_proj, D, D);
    }

    float* out_pos  = (float*)d_out;                       // [B,16,3]
    float* out_feat = out_pos + (size_t)B * 16 * 3;        // [B,16,D]
    float* out_pad  = out_feat + (size_t)B * 16 * D;       // [B,16] (false)

    for (int b = 0; b < B; ++b) {
        extract_kernel<<<(N0 * (3 + D) + 255) / 256, 256, 0, stream>>>(
            pts + (size_t)b * N0 * (3 + D), posA, xA, N0, D);

        float* xc = xA; float* xn = xB;
        float* pc = posA; float* pn = posB;
        int N = N0;
        for (int s = 0; s < 4; ++s) {
            const int M  = N / 4;
            const int MK = M * KNN_K;
            const float r2 = radii[s] * radii[s];

            fps_kernel<<<1, FPS_T, 0, stream>>>(pc, N, M, idx);
            gather_pos_kernel<<<(M * 3 + 255) / 256, 256, 0, stream>>>(pc, idx, pn, M);
            knn_kernel<<<M / 8, 256, 0, stream>>>(pn, pc, M, N, nbr, nd2);

            run_gemm(xc, wt_lin[s], lin_b + (size_t)s * D, nullptr, vbuf, N, D, D, false, stream);
            run_gemm(xc, wt_src[s], nullptr,               nullptr, asrc, N, D, D, false, stream);
            run_gemm(xc, wt_dst[s], nullptr,               idx,     adst, M, D, D, false, stream);

            pos_mlp1_kernel<<<(MK * D + 255) / 256, 256, 0, stream>>>(
                pn, pc, nbr, pos_w1 + (size_t)s * 3 * D, pos_b1 + (size_t)s * D, hidden, M, D);
            run_gemm(hidden, wt_pw2[s], pos_b2 + (size_t)s * D, nullptr, delta, MK, D, D, true, stream);

            attn_input_kernel<<<(MK * D + 255) / 256, 256, 0, stream>>>(
                adst, asrc, delta, nbr, hidden, M, D);
            run_gemm(hidden, wt_attn[s], attn_b + (size_t)s * D, nullptr, alpha, MK, D, D, true, stream);

            softmax_combine_kernel<<<(M * D + 255) / 256, 256, 0, stream>>>(
                alpha, vbuf, delta, nbr, nd2, r2, xn, M, D);
            layernorm_kernel<<<(M + 7) / 8, 256, 0, stream>>>(
                xn, ln_g + (size_t)s * D, ln_b + (size_t)s * D, M, D);

            { float* t = xc; xc = xn; xn = t; }
            { float* t = pc; pc = pn; pn = t; }
            N = M;
        }

        run_gemm(xc, wt_proj, proj_b, nullptr, featb, 16, D, D, false, stream);
        layernorm_kernel<<<2, 256, 0, stream>>>(featb, pln_g, pln_b, 16, D);
        copy_kernel<<<1, 64, 0, stream>>>(pc, out_pos + (size_t)b * 16 * 3, 16 * 3);
        copy_kernel<<<(16 * D + 255) / 256, 256, 0, stream>>>(
            featb, out_feat + (size_t)b * 16 * D, 16 * D);
    }
    zero_kernel<<<1, 64, 0, stream>>>(out_pad, B * 16);
}
